// OPTParallelDroplessMLP_19954418057264
// MI455X (gfx1250) — compile-verified, matrix operand order checked by reading, not run on hardware
//
#include <hip/hip_runtime.h>
#include <hip/hip_bf16.h>
#include <stdint.h>

// ---------------------------------------------------------------------------
// MoE dropless MLP for MI455X (gfx1250, wave32, WMMA, double-buffered
// async-to-LDS staging driven by ASYNCcnt)
// route -> cvt/transpose(bf16) -> gather-GEMM1 (GELU) -> GEMM2 (gate) -> combine
// ---------------------------------------------------------------------------

namespace {
constexpr int T_TOK = 4096;
constexpr int H_DIM = 1024;
constexpr int F_DIM = 2048;
constexpr int E_EXP = 8;
constexpr int K_TOP = 2;
constexpr int R_TOT = T_TOK * K_TOP;   // 8192 routed rows

constexpr int BM = 64;    // rows per block tile
constexpr int BN = 128;   // cols per block tile
constexpr int BK = 32;    // K per WMMA step
constexpr int NTHREADS = 256;  // 8 waves (wave32)
constexpr int LDK = BK + 8;    // LDS K-stride: 80B rows, 16B aligned, conflict-free
constexpr int LDC = BN + 8;    // LDS C-tile stride: 272B rows, 16B aligned

constexpr size_t MB = 1024 * 1024;
// workspace layout (bytes)
constexpr size_t OFF_COUNTS = 0;
constexpr size_t OFF_ROWIDS = 1024;          // E*R ints = 256KB
constexpr size_t OFF_XBF    = 1 * MB;        // x  bf16: T*H   = 8MB
constexpr size_t OFF_W1T    = 16 * MB;       // w1^T bf16 [E][F][H] = 32MB
constexpr size_t OFF_W2T    = 48 * MB;       // w2^T bf16 [E][H][F] = 32MB
constexpr size_t OFF_H      = 80 * MB;       // h  bf16: R*F = 32MB
constexpr size_t OFF_Y      = 112 * MB;      // y  bf16: R*H = 16MB
}

typedef __attribute__((ext_vector_type(4)))  __bf16 v4bf;
typedef __attribute__((ext_vector_type(8)))  __bf16 v8bf;
typedef __attribute__((ext_vector_type(16))) __bf16 v16bf;
typedef __attribute__((ext_vector_type(8)))  float  v8f;
typedef __attribute__((ext_vector_type(4)))  float  v4f;

union BF16Frag { v16bf v; v8bf h[2]; };

__device__ __forceinline__ float gelu_tanh(float x) {
  const float c = 0.7978845608028654f;   // sqrt(2/pi)
  float x3 = x * x * x;
  float t = tanhf(c * (x + 0.044715f * x3));
  return 0.5f * x * (1.0f + t);
}

// --- CDNA5 async global->LDS staging (ASYNCcnt-tracked, 16B per lane) ------
__device__ __forceinline__ void async_ld16(unsigned lds_addr, const void* g) {
  asm volatile("global_load_async_to_lds_b128 %0, %1, off"
               :: "v"(lds_addr), "v"((unsigned long long)(uintptr_t)g)
               : "memory");
}
template <int N>
__device__ __forceinline__ void wait_async() {
#if __has_builtin(__builtin_amdgcn_s_wait_asynccnt)
  __builtin_amdgcn_s_wait_asynccnt(N);
#else
  asm volatile("s_wait_asynccnt %0" :: "i"(N) : "memory");
#endif
}
__device__ __forceinline__ unsigned lds_lo32(const void* p) {
  return (unsigned)(uintptr_t)p;   // LDS aperture: low 32 bits are the LDS offset
}

// ---------------------------------------------------------------------------
__global__ void moe_zero_counts(int* __restrict__ counts) {
  if (threadIdx.x < E_EXP) counts[threadIdx.x] = 0;
}

__global__ void moe_route(const int* __restrict__ top_experts,
                          int* __restrict__ counts,
                          int* __restrict__ rowids) {
  int i = blockIdx.x * blockDim.x + threadIdx.x;
  if (i >= R_TOT) return;
  int e = top_experts[i];                 // flat [T][K]
  int pos = atomicAdd(&counts[e], 1);
  rowids[e * R_TOT + pos] = i;            // row id i = t*K + k (also indexes ew)
}

// --- x fp32 -> bf16 --------------------------------------------------------
__global__ void moe_cvt_x(const float* __restrict__ x, __bf16* __restrict__ xbf) {
  int idx = blockIdx.x * blockDim.x + threadIdx.x;    // vec8 chunk
  const float* p = x + (size_t)idx * 8;
  v4f a0 = *(const v4f*)p;
  v4f a1 = *(const v4f*)(p + 4);
  v8bf t;
#pragma unroll
  for (int i = 0; i < 4; ++i) { t[i] = (__bf16)a0[i]; t[i + 4] = (__bf16)a1[i]; }
  *(v8bf*)(xbf + (size_t)idx * 8) = t;
}

// --- per-expert convert + transpose: in [E][rows][cols] f32 -> out [E][cols][rows] bf16
__global__ void moe_cvt_transpose(const float* __restrict__ in,
                                  __bf16* __restrict__ out,
                                  int rows, int cols) {
  __shared__ __align__(16) __bf16 Ts[32][40];
  const int e  = blockIdx.z;
  const int c0 = blockIdx.x * 32;
  const int r0 = blockIdx.y * 32;
  const int tid = threadIdx.x;
  {
    int r  = tid >> 3;             // 0..31
    int cc = (tid & 7) * 4;        // 0..28
    v4f a = *(const v4f*)(in + (size_t)e * rows * cols + (size_t)(r0 + r) * cols + c0 + cc);
#pragma unroll
    for (int i = 0; i < 4; ++i) Ts[cc + i][r] = (__bf16)a[i];
  }
  __syncthreads();
  {
    int oc  = tid >> 3;            // transposed row (a source column) 0..31
    int orr = (tid & 7) * 4;       // 0..28
    v4bf t;
#pragma unroll
    for (int i = 0; i < 4; ++i) t[i] = Ts[oc][orr + i];
    *(v4bf*)(out + (size_t)e * rows * cols + (size_t)(c0 + oc) * rows + r0 + orr) = t;
  }
}

// ---------------------------------------------------------------------------
// GEMM1: h[r, n] = gelu( sum_k xbf[tok(r), k] * w1t[e, n, k] ), bf16 out
// ---------------------------------------------------------------------------
__global__ __launch_bounds__(NTHREADS)
void moe_gemm1(const __bf16* __restrict__ xbf,
               const __bf16* __restrict__ w1t,   // [E][F][H]
               const int*    __restrict__ counts,
               const int*    __restrict__ rowids,
               __bf16*       __restrict__ hbuf) {
  const int e   = blockIdx.z;
  const int m0  = blockIdx.y * BM;
  const int n0  = blockIdx.x * BN;
  const int cnt = counts[e];
  if (m0 >= cnt) return;   // grid provisioned for worst-case routing skew

  __shared__ __align__(16) __bf16 As[2][BM][LDK];   // [buf][m][k]
  __shared__ __align__(16) __bf16 Bs[2][BN][LDK];   // [buf][n][k]
  __shared__ __align__(16) __bf16 Cs[BM][LDC];      // epilogue transpose tile
  __shared__ int rid[BM];

  const int tid = threadIdx.x;
  if (tid < BM) {
    int r = m0 + tid;
    rid[tid] = (r < cnt) ? rowids[e * R_TOT + r] : -1;
  }
  __syncthreads();

  // A: one 16B chunk / thread  (64 rows x 4 chunks)
  const int arow = tid >> 2;
  const int akc  = (tid & 3) * 8;
  const int aRid = rid[arow];
  const __bf16* aptr = xbf + (size_t)((aRid >= 0 ? aRid : 0) / K_TOP) * H_DIM + akc;
  // B: two 16B chunks / thread (128 rows x 4 chunks)
  const int brow = tid >> 1;
  const int bkc  = (tid & 1) * 16;
  const __bf16* bptr = w1t + (size_t)e * F_DIM * H_DIM + (size_t)(n0 + brow) * H_DIM + bkc;

  const unsigned aLds[2] = { lds_lo32(&As[0][arow][akc]), lds_lo32(&As[1][arow][akc]) };
  const unsigned bLds[2] = { lds_lo32(&Bs[0][brow][bkc]), lds_lo32(&Bs[1][brow][bkc]) };

  // wave tiling: 2x4 waves, each wave a 32x32 tile = 2x2 WMMA tiles
  const int lane = tid & 31;
  const int wv   = tid >> 5;
  const int wm   = (wv >> 2) * 32;
  const int wn   = (wv & 3) * 32;
  const int lrow = lane & 15;
  const int kb   = (lane < 16) ? 0 : 8;

  v8f acc[2][2] = {};

  constexpr int NT = H_DIM / BK;
  // prologue: tile 0 -> buffer 0
  async_ld16(aLds[0], aptr);
  async_ld16(bLds[0], bptr);
  async_ld16(bLds[0] + 16, bptr + 8);

#pragma unroll 2
  for (int it = 0; it < NT; ++it) {
    const int cb = it & 1;
    const int nb = (it + 1) & 1;
    if (it + 1 < NT) {          // prefetch tile it+1 into the other buffer
      const int kn = (it + 1) * BK;
      async_ld16(aLds[nb], aptr + kn);
      async_ld16(bLds[nb], bptr + kn);
      async_ld16(bLds[nb] + 16, bptr + kn + 8);
      wait_async<3>();          // tile `it` landed; tile `it+1` still in flight
    } else {
      wait_async<0>();
    }
    __syncthreads();            // whole tile `it` visible to all waves

    BF16Frag afrag[2], bfrag[2];
#pragma unroll
    for (int i = 0; i < 2; ++i) {
      const __bf16* ap = &As[cb][wm + i * 16 + lrow][kb];
      afrag[i].h[0] = *(const v8bf*)ap;
      afrag[i].h[1] = *(const v8bf*)(ap + 16);
      const __bf16* bp = &Bs[cb][wn + i * 16 + lrow][kb];
      bfrag[i].h[0] = *(const v8bf*)bp;
      bfrag[i].h[1] = *(const v8bf*)(bp + 16);
    }
#pragma unroll
    for (int i = 0; i < 2; ++i)
#pragma unroll
      for (int j = 0; j < 2; ++j)
        acc[i][j] = __builtin_amdgcn_wmma_f32_16x16x32_bf16(
            false, afrag[i].v, false, bfrag[j].v, (short)0, acc[i][j], false, false);
    __syncthreads();            // buffer `cb` free for the prefetch 2 steps ahead
  }

  // epilogue: GELU -> bf16 -> LDS transpose tile -> coalesced b128 stores
  const int mhalf = (lane < 16) ? 0 : 8;
  const int ncol  = lane & 15;
#pragma unroll
  for (int i = 0; i < 2; ++i)
#pragma unroll
    for (int v = 0; v < 8; ++v) {
      int m = wm + i * 16 + mhalf + v;
#pragma unroll
      for (int j = 0; j < 2; ++j)
        Cs[m][wn + j * 16 + ncol] = (__bf16)gelu_tanh(acc[i][j][v]);
    }
  __syncthreads();
#pragma unroll
  for (int it = 0; it < 4; ++it) {
    int chunk = tid + it * NTHREADS;     // 1024 chunks of 8 bf16
    int m  = chunk >> 4;
    int cc = (chunk & 15) * 8;
    int r  = rid[m];
    if (r >= 0)
      *(v8bf*)(hbuf + (size_t)r * F_DIM + n0 + cc) = *(const v8bf*)&Cs[m][cc];
  }
}

// ---------------------------------------------------------------------------
// GEMM2: y[r, n] = gate[r] * sum_k h[r, k] * w2t[e, n, k], bf16 out
// ---------------------------------------------------------------------------
__global__ __launch_bounds__(NTHREADS)
void moe_gemm2(const __bf16* __restrict__ hbuf,
               const __bf16* __restrict__ w2t,   // [E][H][F]
               const int*    __restrict__ counts,
               const int*    __restrict__ rowids,
               const float*  __restrict__ expert_weights,
               __bf16*       __restrict__ ybuf) {
  const int e   = blockIdx.z;
  const int m0  = blockIdx.y * BM;
  const int n0  = blockIdx.x * BN;
  const int cnt = counts[e];
  if (m0 >= cnt) return;

  __shared__ __align__(16) __bf16 As[2][BM][LDK];
  __shared__ __align__(16) __bf16 Bs[2][BN][LDK];
  __shared__ __align__(16) __bf16 Cs[BM][LDC];
  __shared__ int   rid[BM];
  __shared__ float wgt[BM];

  const int tid = threadIdx.x;
  if (tid < BM) {
    int r = m0 + tid;
    int id = (r < cnt) ? rowids[e * R_TOT + r] : -1;
    rid[tid] = id;
    wgt[tid] = (id >= 0) ? expert_weights[id] : 0.0f;
  }
  __syncthreads();

  const int arow = tid >> 2;
  const int akc  = (tid & 3) * 8;
  const int aRid = rid[arow];
  const __bf16* aptr = hbuf + (size_t)(aRid >= 0 ? aRid : 0) * F_DIM + akc;
  const int brow = tid >> 1;
  const int bkc  = (tid & 1) * 16;
  const __bf16* bptr = w2t + (size_t)e * H_DIM * F_DIM + (size_t)(n0 + brow) * F_DIM + bkc;

  const unsigned aLds[2] = { lds_lo32(&As[0][arow][akc]), lds_lo32(&As[1][arow][akc]) };
  const unsigned bLds[2] = { lds_lo32(&Bs[0][brow][bkc]), lds_lo32(&Bs[1][brow][bkc]) };

  const int lane = tid & 31;
  const int wv   = tid >> 5;
  const int wm   = (wv >> 2) * 32;
  const int wn   = (wv & 3) * 32;
  const int lrow = lane & 15;
  const int kb   = (lane < 16) ? 0 : 8;

  v8f acc[2][2] = {};

  constexpr int NT = F_DIM / BK;
  async_ld16(aLds[0], aptr);
  async_ld16(bLds[0], bptr);
  async_ld16(bLds[0] + 16, bptr + 8);

#pragma unroll 2
  for (int it = 0; it < NT; ++it) {
    const int cb = it & 1;
    const int nb = (it + 1) & 1;
    if (it + 1 < NT) {
      const int kn = (it + 1) * BK;
      async_ld16(aLds[nb], aptr + kn);
      async_ld16(bLds[nb], bptr + kn);
      async_ld16(bLds[nb] + 16, bptr + kn + 8);
      wait_async<3>();
    } else {
      wait_async<0>();
    }
    __syncthreads();

    BF16Frag afrag[2], bfrag[2];
#pragma unroll
    for (int i = 0; i < 2; ++i) {
      const __bf16* ap = &As[cb][wm + i * 16 + lrow][kb];
      afrag[i].h[0] = *(const v8bf*)ap;
      afrag[i].h[1] = *(const v8bf*)(ap + 16);
      const __bf16* bp = &Bs[cb][wn + i * 16 + lrow][kb];
      bfrag[i].h[0] = *(const v8bf*)bp;
      bfrag[i].h[1] = *(const v8bf*)(bp + 16);
    }
#pragma unroll
    for (int i = 0; i < 2; ++i)
#pragma unroll
      for (int j = 0; j < 2; ++j)
        acc[i][j] = __builtin_amdgcn_wmma_f32_16x16x32_bf16(
            false, afrag[i].v, false, bfrag[j].v, (short)0, acc[i][j], false, false);
    __syncthreads();
  }

  const int mhalf = (lane < 16) ? 0 : 8;
  const int ncol  = lane & 15;
#pragma unroll
  for (int i = 0; i < 2; ++i)
#pragma unroll
    for (int v = 0; v < 8; ++v) {
      int m = wm + i * 16 + mhalf + v;
      float g = wgt[m];
#pragma unroll
      for (int j = 0; j < 2; ++j)
        Cs[m][wn + j * 16 + ncol] = (__bf16)(g * acc[i][j][v]);
    }
  __syncthreads();
#pragma unroll
  for (int it = 0; it < 4; ++it) {
    int chunk = tid + it * NTHREADS;
    int m  = chunk >> 4;
    int cc = (chunk & 15) * 8;
    int r  = rid[m];
    if (r >= 0)
      *(v8bf*)(ybuf + (size_t)r * H_DIM + n0 + cc) = *(const v8bf*)&Cs[m][cc];
  }
}

// ---------------------------------------------------------------------------
// combine: out[t, :] = y[t*K+0, :] + y[t*K+1, :]   (each y row written once)
// ---------------------------------------------------------------------------
__global__ void moe_combine(const __bf16* __restrict__ y, float* __restrict__ out) {
  int idx = blockIdx.x * blockDim.x + threadIdx.x;     // vec8 chunk id over T*H
  int t  = idx / (H_DIM / 8);
  int hc = (idx % (H_DIM / 8)) * 8;
  v8bf y0 = *(const v8bf*)(y + (size_t)(2 * t)     * H_DIM + hc);
  v8bf y1 = *(const v8bf*)(y + (size_t)(2 * t + 1) * H_DIM + hc);
  v4f o0, o1;
#pragma unroll
  for (int i = 0; i < 4; ++i) {
    o0[i] = (float)y0[i]     + (float)y1[i];
    o1[i] = (float)y0[i + 4] + (float)y1[i + 4];
  }
  float* orow = out + (size_t)t * H_DIM + hc;
  *(v4f*)orow       = o0;
  *(v4f*)(orow + 4) = o1;
}

// ---------------------------------------------------------------------------
extern "C" void kernel_launch(void* const* d_in, const int* in_sizes, int n_in,
                              void* d_out, int out_size, void* d_ws, size_t ws_size,
                              hipStream_t stream) {
  const float* x  = (const float*)d_in[0];
  const float* ew = (const float*)d_in[1];
  const int*   te = (const int*)d_in[2];
  const float* w1 = (const float*)d_in[3];
  const float* w2 = (const float*)d_in[4];
  float* out = (float*)d_out;

  char* ws = (char*)d_ws;
  int*    counts = (int*)(ws + OFF_COUNTS);
  int*    rowids = (int*)(ws + OFF_ROWIDS);
  __bf16* xbf    = (__bf16*)(ws + OFF_XBF);
  __bf16* w1t    = (__bf16*)(ws + OFF_W1T);
  __bf16* w2t    = (__bf16*)(ws + OFF_W2T);
  __bf16* hbuf   = (__bf16*)(ws + OFF_H);
  __bf16* ybuf   = (__bf16*)(ws + OFF_Y);

  moe_zero_counts<<<1, 32, 0, stream>>>(counts);
  moe_route<<<R_TOT / 256, 256, 0, stream>>>(te, counts, rowids);
  moe_cvt_x<<<(T_TOK * H_DIM / 8) / 256, 256, 0, stream>>>(x, xbf);
  moe_cvt_transpose<<<dim3(F_DIM / 32, H_DIM / 32, E_EXP), 256, 0, stream>>>(w1, w1t, H_DIM, F_DIM);
  moe_cvt_transpose<<<dim3(H_DIM / 32, F_DIM / 32, E_EXP), 256, 0, stream>>>(w2, w2t, F_DIM, H_DIM);
  moe_gemm1<<<dim3(F_DIM / BN, R_TOT / BM, E_EXP), NTHREADS, 0, stream>>>(
      xbf, w1t, counts, rowids, hbuf);
  moe_gemm2<<<dim3(H_DIM / BN, R_TOT / BM, E_EXP), NTHREADS, 0, stream>>>(
      hbuf, w2t, counts, rowids, ew, ybuf);
  moe_combine<<<(T_TOK * H_DIM / 8) / 256, 256, 0, stream>>>(ybuf, out);
}
